// SioConvLayer_52922587021618
// MI455X (gfx1250) — compile-verified, matrix operand order checked by reading, not run on hardware
//
#include <hip/hip_runtime.h>
#include <hip/hip_bf16.h>
#include <stdint.h>

#define B_      4
#define L_      1024
#define DIM_    1024
#define H_      8
#define I_      64
#define HI2_    1024
#define NQKG    3072
#define CHUNK   64
#define NCH     16
#define A_SCALE 0.99999f
#define GN_EPS  1e-5f

typedef __attribute__((ext_vector_type(16))) __bf16 v16bf;
typedef __attribute__((ext_vector_type(8)))  float  v8f;
typedef __attribute__((ext_vector_type(4)))  unsigned int u32x4_t;
typedef __attribute__((ext_vector_type(8)))  int          i32x8_t;
typedef __attribute__((ext_vector_type(4)))  int          i32x4_t;

union FragBF { uint4 u[2]; v16bf v; };

__device__ __forceinline__ uint16_t f2bf(float f){
  uint32_t u = __float_as_uint(f);
  u += 0x7fffu + ((u >> 16) & 1u);
  return (uint16_t)(u >> 16);
}

// ---- TDM: async 2-D bf16 tile load Global->LDS with row padding (stride 80B for 64B rows) ----
// D# group0: count=1, lds_addr, global_addr, type=2("image")
// D# group1: data_size=2B, pad_enable, pad_interval=16 DWORDs, pad_amount=4 DWORDs,
//            tensor_dim0/1, tile_dim0/1, tensor_dim0_stride
__device__ __forceinline__ void tdm_load_tile_2d(uint32_t lds_off, const uint16_t* gptr,
                                                 uint32_t tensor_d0, uint32_t tensor_d1,
                                                 uint32_t tile_d0, uint32_t tile_d1,
                                                 uint32_t stride_elems){
  uint64_t ga = (uint64_t)(uintptr_t)gptr;
  u32x4_t g0;
  g0[0] = 1u;                                                  // count=1, user mode
  g0[1] = lds_off;                                             // lds_addr [63:32]
  g0[2] = (uint32_t)ga;                                        // global_addr [95:64]
  g0[3] = ((uint32_t)(ga >> 32) & 0x01ffffffu) | (2u << 30);   // addr[56:32] | type=2
  i32x8_t g1;
  g1[0] = (int)((1u << 16) | (1u << 20) | (3u << 22) | (3u << 25)); // ds=2B,pad,int=16dw,amt=4dw
  g1[1] = (int)((tensor_d0 & 0xffffu) << 16);                  // tensor_dim0[15:0] @ bits63:48
  g1[2] = (int)((tensor_d0 >> 16) | ((tensor_d1 & 0xffffu) << 16));
  g1[3] = (int)((tensor_d1 >> 16) | (tile_d0 << 16));          // tile_dim0 @ bits127:112
  g1[4] = (int)(tile_d1 & 0xffffu);                            // tile_dim1; tile_dim2=0
  g1[5] = (int)stride_elems;                                   // tensor_dim0_stride[31:0]
  g1[6] = 0;
  g1[7] = 0;
  i32x4_t gz4 = {0, 0, 0, 0};
  i32x8_t gz8 = {0, 0, 0, 0, 0, 0, 0, 0};
  __builtin_amdgcn_tensor_load_to_lds(g0, g1, gz4, gz4, gz8, 0);
}

// ---------------- bf16 convert (x) ----------------
__launch_bounds__(256)
__global__ void cvt_bf16_kernel(const float* __restrict__ src, uint16_t* __restrict__ dst, int n4){
  int i = blockIdx.x * 256 + threadIdx.x;
  if (i < n4){
    float4 v = ((const float4*)src)[i];
    uint2 o;
    o.x = (uint32_t)f2bf(v.x) | ((uint32_t)f2bf(v.y) << 16);
    o.y = (uint32_t)f2bf(v.z) | ((uint32_t)f2bf(v.w) << 16);
    ((uint2*)dst)[i] = o;
  }
}

// ---------------- transpose + convert weights: src (K x N) f32 -> dst (N x K) bf16 ----------------
__launch_bounds__(256)
__global__ void transpose_cvt_kernel(const float* __restrict__ src, uint16_t* __restrict__ dst,
                                     int K, int N){
  __shared__ float tile[32][33];
  int n0 = blockIdx.x * 32, k0 = blockIdx.y * 32;
  for (int j = threadIdx.y; j < 32; j += 8)
    tile[j][threadIdx.x] = src[(size_t)(k0 + j) * N + n0 + threadIdx.x];
  __syncthreads();
  for (int j = threadIdx.y; j < 32; j += 8)
    dst[(size_t)(n0 + j) * K + k0 + threadIdx.x] = f2bf(tile[threadIdx.x][j]);
}

// ---------------- bias pack for fused qkg GEMM ----------------
__launch_bounds__(256)
__global__ void biaspack_kernel(const float* __restrict__ bq, const float* __restrict__ bk,
                                const float* __restrict__ bg, float* __restrict__ dst){
  int i = blockIdx.x * 256 + threadIdx.x;
  if (i < NQKG)
    dst[i] = (i < 1024) ? bq[i] : (i < 2048 ? bk[i - 1024] : bg[i - 2048]);
}

// ---------------- bf16 WMMA GEMM: C(MxN,f32) = A(MxK,bf16) @ Bt(NxK,bf16)^T + bias ----------------
// Block tile 128(M) x 256(N), BK=32, double-buffered LDS filled by TDM async tensor loads.
// 8 waves as 2(M) x 4(N); each wave computes 4x4 WMMA 16x16 tiles (64x64).
__launch_bounds__(256)
__global__ void gemm_bf16_kernel(const uint16_t* __restrict__ A, const uint16_t* __restrict__ Bt,
                                 const float* __restrict__ bias, float* __restrict__ C,
                                 int M, int N, int K){
  __shared__ uint16_t sA[2][128 * 40];   // 32 bf16/row + 8 pad (TDM pads to 80B stride)
  __shared__ uint16_t sB[2][256 * 40];
  const int tid  = threadIdx.x;
  const int wave = tid >> 5, lane = tid & 31;
  const int m0 = blockIdx.y * 128, n0 = blockIdx.x * 256;
  const int wm = (wave >> 2) * 64, wn = (wave & 3) * 64;
  const int half = lane >> 4, l15 = lane & 15;

  v8f acc[4][4];
  v8f zero = {0.f,0.f,0.f,0.f,0.f,0.f,0.f,0.f};
  for (int mi = 0; mi < 4; ++mi)
    for (int ni = 0; ni < 4; ++ni) acc[mi][ni] = zero;

  const int nk = K >> 5;
  if (tid == 0){
    tdm_load_tile_2d((uint32_t)(uintptr_t)&sA[0][0], A + (size_t)m0 * K,
                     (uint32_t)K, (uint32_t)M, 32u, 128u, (uint32_t)K);
    tdm_load_tile_2d((uint32_t)(uintptr_t)&sB[0][0], Bt + (size_t)n0 * K,
                     (uint32_t)K, (uint32_t)N, 32u, 256u, (uint32_t)K);
  }

  for (int it = 0; it < nk; ++it){
    const int cur = it & 1;
    if (tid == 0){
      if (it + 1 < nk){
        int kt = (it + 1) << 5;
        tdm_load_tile_2d((uint32_t)(uintptr_t)&sA[cur ^ 1][0], A + (size_t)m0 * K + kt,
                         (uint32_t)K, (uint32_t)M, 32u, 128u, (uint32_t)K);
        tdm_load_tile_2d((uint32_t)(uintptr_t)&sB[cur ^ 1][0], Bt + (size_t)n0 * K + kt,
                         (uint32_t)K, (uint32_t)N, 32u, 256u, (uint32_t)K);
        __builtin_amdgcn_s_wait_tensorcnt(2);   // retire current buffer's pair, keep next in flight
      } else {
        __builtin_amdgcn_s_wait_tensorcnt(0);
      }
    }
    __syncthreads();                            // current buffer visible to all waves

    FragBF fa[4], fb[4];
    for (int mi = 0; mi < 4; ++mi){
      int r = wm + mi * 16 + l15;               // lanes 0-15: K0-7 & 16-23; lanes 16-31: K8-15 & 24-31
      fa[mi].u[0] = *(const uint4*)&sA[cur][r * 40 + half * 8];
      fa[mi].u[1] = *(const uint4*)&sA[cur][r * 40 + 16 + half * 8];
    }
    for (int ni = 0; ni < 4; ++ni){
      int c = wn + ni * 16 + l15;
      fb[ni].u[0] = *(const uint4*)&sB[cur][c * 40 + half * 8];
      fb[ni].u[1] = *(const uint4*)&sB[cur][c * 40 + 16 + half * 8];
    }
    for (int mi = 0; mi < 4; ++mi)
      for (int ni = 0; ni < 4; ++ni)
        acc[mi][ni] = __builtin_amdgcn_wmma_f32_16x16x32_bf16(
            false, fa[mi].v, false, fb[ni].v, (short)0, acc[mi][ni], false, false);
    __syncthreads();                            // buffer free before TDM reuses it next iteration
  }

  for (int mi = 0; mi < 4; ++mi)
    for (int ni = 0; ni < 4; ++ni){
      int col = n0 + wn + ni * 16 + l15;
      float bv = bias ? bias[col] : 0.f;
      for (int r = 0; r < 8; ++r){
        int rr = m0 + wm + mi * 16 + half * 8 + r;   // C/D layout: lanes 16-31 hold M=r+8
        C[(size_t)rr * N + col] = acc[mi][ni][r] + bv;
      }
    }
}

// ---------------- a_ang = x @ Wa + ba   (4096 x 1024) @ (1024 x 8) ----------------
__launch_bounds__(256)
__global__ void aang_kernel(const float* __restrict__ x, const float* __restrict__ Wa,
                            const float* __restrict__ ba, float* __restrict__ out){
  int wave = threadIdx.x >> 5, lane = threadIdx.x & 31;
  int m = blockIdx.x * 8 + wave;
  float acc[8];
  for (int h = 0; h < 8; ++h) acc[h] = 0.f;
  const float* xr = x + (size_t)m * DIM_;
  for (int k = lane; k < DIM_; k += 32){
    float xv = xr[k];
    const float4* w = (const float4*)&Wa[k * 8];
    float4 w0 = w[0], w1 = w[1];
    acc[0] += xv * w0.x; acc[1] += xv * w0.y; acc[2] += xv * w0.z; acc[3] += xv * w0.w;
    acc[4] += xv * w1.x; acc[5] += xv * w1.y; acc[6] += xv * w1.z; acc[7] += xv * w1.w;
  }
  for (int h = 0; h < 8; ++h)
    for (int s = 16; s > 0; s >>= 1) acc[h] += __shfl_xor(acc[h], s, 32);
  if (lane == 0){
    float4 o0 = {acc[0]+ba[0], acc[1]+ba[1], acc[2]+ba[2], acc[3]+ba[3]};
    float4 o1 = {acc[4]+ba[4], acc[5]+ba[5], acc[6]+ba[6], acc[7]+ba[7]};
    ((float4*)&out[(size_t)m * 8])[0] = o0;
    ((float4*)&out[(size_t)m * 8])[1] = o1;
  }
}

// ---------------- scan phase 1: per-chunk local scan h = a*h + k_rot, plus prefix products ----------
__launch_bounds__(256)
__global__ void scan_local_kernel(const float* __restrict__ qkg, const float* __restrict__ aang,
                                  const float* __restrict__ p,
                                  float* __restrict__ hloc, float* __restrict__ ppbuf){
  int g = blockIdx.x * 8 + (threadIdx.x >> 5);   // 0..511 = (b,h,chunk)
  int lane = threadIdx.x & 31;
  int b = g >> 7, h = (g >> 4) & 7, c = g & 15;
  float p0 = p[h * I_ + 2 * lane], p1 = p[h * I_ + 2 * lane + 1];
  float hr0 = 0.f, hi0 = 0.f, hr1 = 0.f, hi1 = 0.f;
  float ppr = 1.f, ppi = 0.f;
  int l0 = c * CHUNK;
  for (int t = 0; t < CHUNK; ++t){
    int l = l0 + t;
    size_t row = (size_t)b * L_ + l;
    float ang = aang[row * 8 + h];
    float sa, ca; __sincosf(ang, &sa, &ca);
    float ar = ca * A_SCALE, ai = sa * A_SCALE;
    float npr = ppr * ar - ppi * ai, npi = ppr * ai + ppi * ar;   // pp *= a
    ppr = npr; ppi = npi;
    float4 kv = *(const float4*)&qkg[row * NQKG + HI2_ + h * 128 + 4 * lane];
    float s0, c0, s1, c1;
    __sincosf((float)l * p0, &s0, &c0);
    __sincosf((float)l * p1, &s1, &c1);
    float kr0 = kv.x * c0 + kv.y * s0, ki0 = kv.y * c0 - kv.x * s0;  // k * conj(rot)
    float kr1 = kv.z * c1 + kv.w * s1, ki1 = kv.w * c1 - kv.z * s1;
    float tr0 = ar * hr0 - ai * hi0 + kr0, ti0 = ar * hi0 + ai * hr0 + ki0;
    float tr1 = ar * hr1 - ai * hi1 + kr1, ti1 = ar * hi1 + ai * hr1 + ki1;
    hr0 = tr0; hi0 = ti0; hr1 = tr1; hi1 = ti1;
    float4 hv = {hr0, hi0, hr1, hi1};
    *(float4*)&hloc[row * HI2_ + h * 128 + 4 * lane] = hv;
    if (lane == 0){
      float2 pv = {ppr, ppi};
      *(float2*)&ppbuf[(size_t)((b * 8 + h) * L_ + l) * 2] = pv;
    }
  }
}

// ---------------- scan phase 2: serial carry combine across 16 chunks ----------------
__launch_bounds__(256)
__global__ void scan_carry_kernel(const float* __restrict__ hloc, const float* __restrict__ ppbuf,
                                  const float* __restrict__ hidden, const float* __restrict__ p,
                                  float* __restrict__ carry){
  int b = blockIdx.x;
  int h = threadIdx.x >> 5, lane = threadIdx.x & 31;
  float p0 = p[h * I_ + 2 * lane], p1 = p[h * I_ + 2 * lane + 1];
  float s0, c0, s1, c1;
  __sincosf(p0, &s0, &c0); __sincosf(p1, &s1, &c1);
  float4 hd = *(const float4*)&hidden[(size_t)((b * 8 + h) * I_ + 2 * lane) * 2];
  float cr0 = hd.x * c0 - hd.y * s0, ci0 = hd.x * s0 + hd.y * c0;   // hidden * exp(i p)
  float cr1 = hd.z * c1 - hd.w * s1, ci1 = hd.z * s1 + hd.w * c1;
  for (int c = 0; c < NCH; ++c){
    float4 cv = {cr0, ci0, cr1, ci1};
    *(float4*)&carry[(size_t)(((b * 8 + h) * NCH + c) * 128) + 4 * lane] = cv;
    int lastl = c * CHUNK + CHUNK - 1;
    float2 P = *(const float2*)&ppbuf[(size_t)((b * 8 + h) * L_ + lastl) * 2];
    size_t row = (size_t)b * L_ + lastl;
    float4 Lc = *(const float4*)&hloc[row * HI2_ + h * 128 + 4 * lane];
    float nr0 = P.x * cr0 - P.y * ci0 + Lc.x, ni0 = P.x * ci0 + P.y * cr0 + Lc.y;
    float nr1 = P.x * cr1 - P.y * ci1 + Lc.z, ni1 = P.x * ci1 + P.y * cr1 + Lc.w;
    cr0 = nr0; ci0 = ni0; cr1 = nr1; ci1 = ni1;
  }
}

// ---------------- scan phase 3: fix-up + q*rot + groupnorm + SiLU gate + bf16 pack ---------------
__launch_bounds__(256)
__global__ void fuse_out_kernel(const float* __restrict__ qkg, const float* __restrict__ hloc,
                                const float* __restrict__ ppbuf, const float* __restrict__ carry,
                                const float* __restrict__ p, const float* __restrict__ gnw,
                                const float* __restrict__ gnb, uint16_t* __restrict__ hng,
                                float* __restrict__ hidden_out){
  int bl = blockIdx.x;
  int b = bl >> 10, l = bl & 1023;
  int h = threadIdx.x >> 5, lane = threadIdx.x & 31;
  int chunk = l >> 6;
  size_t row = (size_t)b * L_ + l;
  float2 pp = *(const float2*)&ppbuf[(size_t)((b * 8 + h) * L_ + l) * 2];
  float4 cv = *(const float4*)&carry[(size_t)(((b * 8 + h) * NCH + chunk) * 128) + 4 * lane];
  float4 hl = *(const float4*)&hloc[row * HI2_ + h * 128 + 4 * lane];
  float hr0 = pp.x * cv.x - pp.y * cv.y + hl.x;
  float hi0 = pp.x * cv.y + pp.y * cv.x + hl.y;
  float hr1 = pp.x * cv.z - pp.y * cv.w + hl.z;
  float hi1 = pp.x * cv.w + pp.y * cv.z + hl.w;
  if (l == L_ - 1){
    float4 hv = {hr0, hi0, hr1, hi1};
    *(float4*)&hidden_out[(size_t)((b * 8 + h) * I_ + 2 * lane) * 2] = hv;
  }
  float p0 = p[h * I_ + 2 * lane], p1 = p[h * I_ + 2 * lane + 1];
  float s0, c0, s1, c1;
  __sincosf((float)l * p0, &s0, &c0);
  __sincosf((float)l * p1, &s1, &c1);
  float4 qv = *(const float4*)&qkg[row * NQKG + h * 128 + 4 * lane];
  float qr0 = qv.x * c0 - qv.y * s0, qi0 = qv.x * s0 + qv.y * c0;   // q * rot
  float qr1 = qv.z * c1 - qv.w * s1, qi1 = qv.z * s1 + qv.w * c1;
  float xr0 = qr0 * hr0 - qi0 * hi0, xi0 = qr0 * hi0 + qi0 * hr0;
  float xr1 = qr1 * hr1 - qi1 * hi1, xi1 = qr1 * hi1 + qi1 * hr1;
  float s  = xr0 + xi0 + xr1 + xi1;
  float sq = xr0*xr0 + xi0*xi0 + xr1*xr1 + xi1*xi1;
  for (int m = 16; m > 0; m >>= 1){ s += __shfl_xor(s, m, 32); sq += __shfl_xor(sq, m, 32); }
  float mean = s * (1.f / 128.f);
  float var  = sq * (1.f / 128.f) - mean * mean;
  float rstd = rsqrtf(var + GN_EPS);
  float gw = gnw[h], gb = gnb[h];
  float n0 = (xr0 - mean) * rstd * gw + gb;
  float n1 = (xi0 - mean) * rstd * gw + gb;
  float n2 = (xr1 - mean) * rstd * gw + gb;
  float n3 = (xi1 - mean) * rstd * gw + gb;
  float4 gv = *(const float4*)&qkg[row * NQKG + 2048 + h * 128 + 4 * lane];
  n0 *= gv.x / (1.f + __expf(-gv.x));
  n1 *= gv.y / (1.f + __expf(-gv.y));
  n2 *= gv.z / (1.f + __expf(-gv.z));
  n3 *= gv.w / (1.f + __expf(-gv.w));
  uint2 o;
  o.x = (uint32_t)f2bf(n0) | ((uint32_t)f2bf(n1) << 16);
  o.y = (uint32_t)f2bf(n2) | ((uint32_t)f2bf(n3) << 16);
  *(uint2*)&hng[row * HI2_ + h * 128 + 4 * lane] = o;
}

extern "C" void kernel_launch(void* const* d_in, const int* in_sizes, int n_in,
                              void* d_out, int out_size, void* d_ws, size_t ws_size,
                              hipStream_t stream) {
  (void)in_sizes; (void)n_in; (void)out_size; (void)ws_size;
  const float* x      = (const float*)d_in[0];
  const float* hidden = (const float*)d_in[1];
  const float* Wq     = (const float*)d_in[2];
  const float* bq     = (const float*)d_in[3];
  const float* Wk     = (const float*)d_in[4];
  const float* bk     = (const float*)d_in[5];
  const float* Wa     = (const float*)d_in[6];
  const float* ba     = (const float*)d_in[7];
  const float* Wg     = (const float*)d_in[8];
  const float* bg     = (const float*)d_in[9];
  const float* Wy     = (const float*)d_in[10];
  const float* by     = (const float*)d_in[11];
  const float* p      = (const float*)d_in[12];
  const float* gnw    = (const float*)d_in[13];
  const float* gnb    = (const float*)d_in[14];
  float* out = (float*)d_out;

  uint8_t* ws = (uint8_t*)d_ws;
  size_t off = 0;
  auto alloc = [&](size_t bytes) -> void* {
    void* ptr = ws + off;
    off = (off + bytes + 255) & ~(size_t)255;
    return ptr;
  };
  uint16_t* xh      = (uint16_t*)alloc((size_t)4096 * 1024 * 2);   // x in bf16
  uint16_t* WqkgT   = (uint16_t*)alloc((size_t)3072 * 1024 * 2);   // [Wq|Wk|Wg]^T bf16
  uint16_t* WyT     = (uint16_t*)alloc((size_t)1024 * 1024 * 2);   // Wy^T bf16
  float*    biasqkg = (float*)   alloc((size_t)3072 * 4);
  float*    qkg     = (float*)   alloc((size_t)4096 * 3072 * 4);   // fused q|k|g f32
  float*    aang    = (float*)   alloc((size_t)4096 * 8 * 4);
  float*    hloc    = (float*)   alloc((size_t)4096 * 1024 * 4);   // chunk-local scans
  float*    ppbuf   = (float*)   alloc((size_t)32 * 1024 * 2 * 4); // prefix decay products
  float*    carry   = (float*)   alloc((size_t)32 * 16 * 128 * 4); // chunk carry states
  uint16_t* hng     = (uint16_t*)alloc((size_t)4096 * 1024 * 2);   // normed+gated bf16

  // pack inputs
  cvt_bf16_kernel<<<4096, 256, 0, stream>>>(x, xh, 4096 * 1024 / 4);
  dim3 tb(32, 8), tg(32, 32);
  transpose_cvt_kernel<<<tg, tb, 0, stream>>>(Wq, WqkgT,                 1024, 1024);
  transpose_cvt_kernel<<<tg, tb, 0, stream>>>(Wk, WqkgT + 1024 * 1024,   1024, 1024);
  transpose_cvt_kernel<<<tg, tb, 0, stream>>>(Wg, WqkgT + 2048 * 1024,   1024, 1024);
  transpose_cvt_kernel<<<tg, tb, 0, stream>>>(Wy, WyT,                   1024, 1024);
  biaspack_kernel<<<12, 256, 0, stream>>>(bq, bk, bg, biasqkg);

  // fused q|k|g projection: (4096x1024) @ (1024x3072)
  gemm_bf16_kernel<<<dim3(3072 / 256, 4096 / 128), 256, 0, stream>>>(
      xh, WqkgT, biasqkg, qkg, 4096, 3072, 1024);

  // a_ang projection (tiny, memory bound)
  aang_kernel<<<512, 256, 0, stream>>>(x, Wa, ba, aang);

  // chunked complex linear-recurrence scan
  scan_local_kernel<<<64, 256, 0, stream>>>(qkg, aang, p, hloc, ppbuf);
  scan_carry_kernel<<<4, 256, 0, stream>>>(hloc, ppbuf, hidden, p, carry);
  fuse_out_kernel<<<4096, 256, 0, stream>>>(qkg, hloc, ppbuf, carry, p, gnw, gnb,
                                            hng, out + (size_t)4096 * 1024);

  // output projection: (4096x1024) @ (1024x1024) + by -> y (f32, front of d_out)
  gemm_bf16_kernel<<<dim3(1024 / 256, 4096 / 128), 256, 0, stream>>>(
      hng, WyT, by, out, 4096, 1024, 1024);
}
